// ResGCN_35914516529585
// MI455X (gfx1250) — matrix operand order repaired and checked by reading, not compiled, not run on hardware
//
#include <hip/hip_runtime.h>
#include <math.h>

// ---------------- problem constants (match reference) ----------------
#define G      256
#define NPG    360
#define NTOT   (G * NPG)        // 92160
#define DEG    16
#define EPG    (NPG * DEG)      // 5760 edges per graph
#define ETOT   (NTOT * DEG)     // 1474560
#define H      5
#define C      30
#define INCH   11
#define NUM_GC 9
#define EPSN   1e-5f

#define MPAD   368              // 360 rounded up to 16 for WMMA tiles (23 tiles)
#define KPAD   32               // C=30 padded to one WMMA K-step
#define FCK    320              // 300 padded to 10 K-steps of 32
#define FCN    64               // 50 padded to 4 N-tiles of 16

typedef __attribute__((ext_vector_type(16))) _Float16 v16h;
typedef __attribute__((ext_vector_type(8)))  float    v8f;

// ---------------- async global -> LDS copy (gfx1250 ASYNCcnt path) ----------
// Each lane copies 16B per issued instruction; whole wave moves 512B/inst with
// no VGPR staging. Caller must async_wait() + __syncthreads() before use.
// Requires 16B-aligned base pointers and bytes % 16 == 0.
__device__ __forceinline__ void g2l_async_b128(void* lds, const void* g, int bytes,
                                               int tid, int nthr) {
#if defined(__gfx1250__)
  unsigned l0 = (unsigned)(size_t)lds;                 // low 32 bits = LDS offset
  unsigned long long g0 = (unsigned long long)(size_t)g;
  for (int o = tid * 16; o < bytes; o += nthr * 16) {
    unsigned la = l0 + (unsigned)o;
    unsigned long long ga = g0 + (unsigned long long)(unsigned)o;
    asm volatile("global_load_async_to_lds_b128 %0, %1, off"
                 :: "v"(la), "v"(ga) : "memory");
  }
#else
  float4* d = (float4*)lds;
  const float4* s = (const float4*)g;
  for (int o = tid; o < bytes / 16; o += nthr) d[o] = s[o];
#endif
}

__device__ __forceinline__ void async_wait() {
#if defined(__gfx1250__)
  asm volatile("s_wait_asynccnt 0x0" ::: "memory");
#endif
}

// ---------------- WMMA fragment helpers (CDNA5 16x16x32 f16) ----------------
// A: M x K row-major (ld elements). ISA 7.12.2: lane m=L&15, half=L>>4;
// VGPR v holds K pair k = (v>>2)*16 + half*8 + (v&3)*2.
__device__ inline v16h wmma_load_a(const _Float16* __restrict__ A, int row0, int ld, int lane) {
  int m = lane & 15, hs = lane >> 4;
  const _Float16* r = A + (size_t)(row0 + m) * ld;
  v16h a;
#pragma unroll
  for (int v = 0; v < 8; ++v) {
    int k = ((v >> 2) << 4) + (hs << 3) + ((v & 3) << 1);
    a[2 * v]     = r[k];
    a[2 * v + 1] = r[k + 1];
  }
  return a;
}

// B: K(=32) x N row-major (ld elements). lane n=L&15, half=L>>4;
// VGPR j holds rows k, k+1 with k = (j>>2)*16 + half*8 + (j&3)*2 for column n.
__device__ inline v16h wmma_load_b(const _Float16* __restrict__ B, int col0, int ld, int lane) {
  int n = col0 + (lane & 15), hs = lane >> 4;
  v16h b;
#pragma unroll
  for (int j = 0; j < 8; ++j) {
    int k = ((j >> 2) << 4) + (hs << 3) + ((j & 3) << 1);
    b[2 * j]     = B[(size_t)k * ld + n];
    b[2 * j + 1] = B[(size_t)(k + 1) * ld + n];
  }
  return b;
}

// D/C f32 16x16: VGPR v -> row v + 8*(lane>>4), col lane&15.
__device__ inline void wmma_store_d(float* __restrict__ O, const v8f& d, int row0, int col0,
                                    int ld, int lane) {
  int n = col0 + (lane & 15), hs = lane >> 4;
#pragma unroll
  for (int v = 0; v < 8; ++v) O[(size_t)(row0 + v + 8 * hs) * ld + n] = d[v];
}

// =====================================================================
// Kernel: EGAT conv + instance norm + pooled block 0, one block per graph.
// Builds per-graph CSR (sorted by dst) and writes it + per-edge weights
// to scratch for reuse by the GraphConv stack.
// =====================================================================
__global__ void __launch_bounds__(256) k_egat(
    const float* __restrict__ x, const int* __restrict__ ei,
    const float* __restrict__ eattr, const float* __restrict__ egat_w,
    const float* __restrict__ att_src, const float* __restrict__ att_dst,
    const float* __restrict__ att_edge,
    float* __restrict__ h_out,        // [NTOT][C]
    float* __restrict__ ew_out,       // [ETOT] sorted by dst within graph
    int* __restrict__ src_out,        // [ETOT] local src, sorted by dst
    int* __restrict__ start_out,      // [G][NPG+1]
    float* __restrict__ pooled)       // [G][300]
{
  __shared__ float sW[INCH][H][C];                    // egat_w
  __shared__ float sAsrc[INCH][H];
  __shared__ float sAdst[INCH][H];
  __shared__ float sAttE[H];
  __shared__ __align__(16) float sX[NPG * INCH];      // node inputs (async)
  __shared__ __align__(16) float sEa[EPG];            // edge_attr   (async)
  __shared__ __align__(16) int   sSrcL[EPG];          // src ids     (async)
  __shared__ __align__(16) int   sDstL[EPG];          // dst ids     (async)
  __shared__ float sSatt[2][NPG][H];                  // 0: src score, 1: dst score
  __shared__ int   sStart[NPG + 1];
  __shared__ int   sPos[NPG];
  __shared__ int   sEsort[EPG];
  __shared__ float sEw[EPG];
  __shared__ float sT[NPG][H][12];                    // alpha-weighted input sums
  __shared__ float shv[NPG][C];
  __shared__ float sMean[C], sInv[C];

  const int g = blockIdx.x, tid = threadIdx.x;
  const int nodeBase = g * NPG, edgeBase = g * EPG;

  // ---- P0: async bulk preloads overlap with zero-init + weight load ----
  g2l_async_b128(sX,    x + (size_t)nodeBase * INCH, NPG * INCH * 4, tid, 256);
  g2l_async_b128(sEa,   eattr + edgeBase,            EPG * 4,        tid, 256);
  g2l_async_b128(sSrcL, ei + edgeBase,               EPG * 4,        tid, 256);
  g2l_async_b128(sDstL, ei + ETOT + edgeBase,        EPG * 4,        tid, 256);

  for (int i = tid; i < INCH * H * C; i += 256) (&sW[0][0][0])[i] = egat_w[i];
  if (tid < H) sAttE[tid] = att_edge[tid];
  for (int i = tid; i < NPG * H * 12; i += 256) (&sT[0][0][0])[i] = 0.f;
  for (int i = tid; i < EPG; i += 256) sEw[i] = 0.f;
  for (int i = tid; i < NPG + 1; i += 256) sStart[i] = 0;
  for (int i = tid; i < NPG; i += 256) sPos[i] = 0;
  async_wait();
  __syncthreads();

  // ---- P1: in-degree count + fold attention vectors through egat_w ----
  for (int e = tid; e < EPG; e += 256) atomicAdd(&sPos[sDstL[e] - nodeBase], 1);
  for (int w = tid; w < INCH * H; w += 256) {
    int i = w / H, h = w - i * H;
    float as = 0.f, ad = 0.f;
    for (int c = 0; c < C; ++c) {
      as += sW[i][h][c] * att_src[h * C + c];
      ad += sW[i][h][c] * att_dst[h * C + c];
    }
    sAsrc[i][h] = as;
    sAdst[i][h] = ad;
  }
  __syncthreads();

  // ---- P2: prefix sum (serial, tiny) + per-node attention scores ----
  if (tid == 0) {
    int acc = 0;
    for (int n = 0; n < NPG; ++n) { sStart[n] = acc; acc += sPos[n]; }
    sStart[NPG] = acc;
  }
  for (int w = tid; w < NPG * H; w += 256) {
    int n = w / H, h = w - n * H;
    float ss = 0.f, sd = 0.f;
    for (int i = 0; i < INCH; ++i) {
      ss += sX[n * INCH + i] * sAsrc[i][h];
      sd += sX[n * INCH + i] * sAdst[i][h];
    }
    sSatt[0][n][h] = ss;
    sSatt[1][n][h] = sd;
  }
  __syncthreads();

  // ---- P3: reset cursors ----
  for (int n = tid; n < NPG; n += 256) sPos[n] = sStart[n];
  __syncthreads();

  // ---- P4: counting-sort placement + export CSR row starts ----
  for (int e = tid; e < EPG; e += 256) {
    int dl = sDstL[e] - nodeBase;
    sEsort[atomicAdd(&sPos[dl], 1)] = e;
  }
  for (int j = tid; j <= NPG; j += 256) start_out[g * (NPG + 1) + j] = sStart[j];
  __syncthreads();

  // ---- P5: per (node, head) streaming softmax + weighted input aggregation ----
  for (int w = tid; w < NPG * H; w += 256) {
    int n = w / H, h = w - n * H;
    int js = sStart[n], je = sStart[n + 1];
    if (js == je) continue;
    float sdn = sSatt[1][n][h], ae = sAttE[h];
    float m = -1e30f;
    for (int j = js; j < je; ++j) {
      int e = sEsort[j];
      int sl = sSrcL[e] - nodeBase;
      float lg = sSatt[0][sl][h] + sdn + sEa[e] * ae;
      lg = lg > 0.f ? lg : 0.2f * lg;
      m = fmaxf(m, lg);
    }
    float denom = 0.f;
    for (int j = js; j < je; ++j) {
      int e = sEsort[j];
      int sl = sSrcL[e] - nodeBase;
      float lg = sSatt[0][sl][h] + sdn + sEa[e] * ae;
      lg = lg > 0.f ? lg : 0.2f * lg;
      denom += expf(lg - m);
    }
    float inv = 1.f / (denom + 1e-16f);
    float t[INCH];
    for (int i = 0; i < INCH; ++i) t[i] = 0.f;
    for (int j = js; j < je; ++j) {
      int e = sEsort[j];
      int sl = sSrcL[e] - nodeBase;
      float lg = sSatt[0][sl][h] + sdn + sEa[e] * ae;
      lg = lg > 0.f ? lg : 0.2f * lg;
      float alpha = expf(lg - m) * inv;
      atomicAdd(&sEw[e], alpha * (1.0f / H));     // ew = alpha.mean(heads)
      for (int i = 0; i < INCH; ++i) t[i] += alpha * sX[sl * INCH + i];
    }
    for (int i = 0; i < INCH; ++i) sT[n][h][i] = t[i];
  }
  __syncthreads();

  // ---- P6: export sorted edge data; dense contraction h0 = (1/H) T . W ----
  for (int j = tid; j < EPG; j += 256) {
    int e = sEsort[j];
    ew_out[edgeBase + j]  = sEw[e];
    src_out[edgeBase + j] = sSrcL[e] - nodeBase;
  }
  for (int w = tid; w < NPG * C; w += 256) {
    int n = w / C, c = w - n * C;
    float acc = 0.f;
    for (int h = 0; h < H; ++h)
      for (int i = 0; i < INCH; ++i) acc += sT[n][h][i] * sW[i][h][c];
    shv[n][c] = acc * (1.0f / H);
  }
  __syncthreads();

  // ---- P7..P9: per-graph instance norm + write + pool block 0 ----
  if (tid < C) {
    float s = 0.f, s2 = 0.f;
    for (int n = 0; n < NPG; ++n) { float v = shv[n][tid]; s += v; s2 += v * v; }
    float mean = s / NPG, var = s2 / NPG - mean * mean;
    sMean[tid] = mean;
    sInv[tid]  = rsqrtf(var + EPSN);
  }
  __syncthreads();
  for (int w = tid; w < NPG * C; w += 256) {
    int n = w / C, c = w - n * C;
    float v = (shv[n][c] - sMean[c]) * sInv[c];
    shv[n][c] = v;
    h_out[(size_t)(nodeBase + n) * C + c] = v;
  }
  __syncthreads();
  if (tid < C) {
    float mx = -1e30f;
    for (int n = 0; n < NPG; ++n) mx = fmaxf(mx, shv[n][tid]);
    pooled[g * 300 + tid] = mx;
  }
}

// =====================================================================
// Kernel: 9 fused GraphConv layers, one block per graph, all-LDS.
// Dense parts (h@W_root, h@W_nbr) run on v_wmma_f32_16x16x32_f16.
// =====================================================================
__global__ void __launch_bounds__(256) k_gc(
    const float* __restrict__ h_in, const float* __restrict__ w_root,
    const float* __restrict__ w_nbr, const float* __restrict__ gc_b,
    const float* __restrict__ ew_s, const int* __restrict__ src_s,
    const int* __restrict__ start_g, float* __restrict__ pooled)
{
  __shared__ __align__(16) float sh[NPG][C];
  __shared__ _Float16 shf[MPAD][KPAD];       // f16 copy of h, zero padded
  __shared__ float    sroot[MPAD][KPAD];     // h @ W_root (f32 accum)
  __shared__ float    shw[MPAD][KPAD];       // h @ W_nbr
  __shared__ _Float16 sWr[KPAD][KPAD];
  __shared__ _Float16 sWn[KPAD][KPAD];
  __shared__ float    sb[C];
  __shared__ int      sStart[NPG + 1];
  __shared__ __align__(16) int   sSrc[EPG];
  __shared__ __align__(16) float sEwl[EPG];
  __shared__ float    sMean[C], sInv[C];

  const int g = blockIdx.x, tid = threadIdx.x;
  const int wave = tid >> 5, lane = tid & 31;
  const int nodeBase = g * NPG, edgeBase = g * EPG;

  // async preload of per-graph state (h, CSR) via the ASYNCcnt path
  g2l_async_b128(&sh[0][0], h_in + (size_t)nodeBase * C, NPG * C * 4, tid, 256);
  g2l_async_b128(sSrc, src_s + edgeBase, EPG * 4, tid, 256);
  g2l_async_b128(sEwl, ew_s + edgeBase,  EPG * 4, tid, 256);
  for (int j = tid; j <= NPG; j += 256) sStart[j] = start_g[g * (NPG + 1) + j];
  async_wait();
  __syncthreads();

  for (int l = 0; l < NUM_GC; ++l) {
    // weights -> f16 padded [K=32][N=32]; h -> f16 padded [368][32]
    const float* wr = w_root + (size_t)l * C * C;
    const float* wn = w_nbr + (size_t)l * C * C;
    if (l + 1 < NUM_GC && tid == 0) {           // warm L2/WGP$ for next layer
      __builtin_prefetch(w_root + (size_t)(l + 1) * C * C, 0, 1);
      __builtin_prefetch(w_nbr + (size_t)(l + 1) * C * C, 0, 1);
    }
    for (int i = tid; i < KPAD * KPAD; i += 256) {
      int k = i >> 5, c = i & 31;
      bool in = (k < C) && (c < C);
      sWr[k][c] = in ? (_Float16)wr[k * C + c] : (_Float16)0.f;
      sWn[k][c] = in ? (_Float16)wn[k * C + c] : (_Float16)0.f;
    }
    if (tid < C) sb[tid] = gc_b[l * C + tid];
    for (int i = tid; i < MPAD * KPAD; i += 256) {
      int r = i >> 5, c = i & 31;
      shf[r][c] = (r < NPG && c < C) ? (_Float16)sh[r][c] : (_Float16)0.f;
    }
    __syncthreads();

    // 23 M-tiles x 2 N-tiles x 2 weight matrices = 92 WMMA tiles over 8 waves
    for (int t = wave; t < 92; t += 8) {
      int mt = t >> 2, r = t & 3;
      int nt = r >> 1, mat = r & 1;
      int row0 = mt * 16, col0 = nt * 16;
      v16h a  = wmma_load_a(&shf[0][0], row0, KPAD, lane);
      v16h bf = wmma_load_b(mat ? &sWn[0][0] : &sWr[0][0], col0, KPAD, lane);
      v8f acc = {};
      acc = __builtin_amdgcn_wmma_f32_16x16x32_f16(false, a, false, bf, (short)0, acc,
                                                   false, false);
      wmma_store_d(mat ? &shw[0][0] : &sroot[0][0], acc, row0, col0, KPAD, lane);
    }
    __syncthreads();

    // combine: h_new = h@Wr + b + sum_{e->n} ew[e] * (h@Wn)[src[e]]
    for (int w = tid; w < NPG * C; w += 256) {
      int n = w / C, c = w - n * C;
      float acc = sroot[n][c] + sb[c];
      int je = sStart[n + 1];
      for (int j = sStart[n]; j < je; ++j) acc += sEwl[j] * shw[sSrc[j]][c];
      sh[n][c] = acc;    // sh only read via shf this layer: safe to overwrite
    }
    __syncthreads();

    // instance norm
    if (tid < C) {
      float s = 0.f, s2 = 0.f;
      for (int n = 0; n < NPG; ++n) { float v = sh[n][tid]; s += v; s2 += v * v; }
      float mean = s / NPG, var = s2 / NPG - mean * mean;
      sMean[tid] = mean;
      sInv[tid]  = rsqrtf(var + EPSN);
    }
    __syncthreads();
    for (int w = tid; w < NPG * C; w += 256) {
      int n = w / C, c = w - n * C;
      sh[n][c] = (sh[n][c] - sMean[c]) * sInv[c];
    }
    __syncthreads();
    if (tid < C) {
      float mx = -1e30f;
      for (int n = 0; n < NPG; ++n) mx = fmaxf(mx, sh[n][tid]);
      pooled[g * 300 + (l + 1) * C + tid] = mx;
    }
    __syncthreads();
  }
}

// =====================================================================
// FC head
// =====================================================================
__global__ void __launch_bounds__(256) k_prep_fc1(const float* __restrict__ fc1_w,
                                                  _Float16* __restrict__ fc1h) {
  int i = blockIdx.x * 256 + threadIdx.x;     // [FCK][FCN]
  if (i >= FCK * FCN) return;
  int k = i / FCN, c = i - k * FCN;
  fc1h[i] = (k < 300 && c < 50) ? (_Float16)fc1_w[k * 50 + c] : (_Float16)0.f;
}

__global__ void __launch_bounds__(256) k_pool_cvt(const float* __restrict__ pooled,
                                                  _Float16* __restrict__ pooledh) {
  int i = blockIdx.x * 256 + threadIdx.x;     // [G][FCK]
  if (i >= G * FCK) return;
  int gi = i / FCK, j = i - gi * FCK;
  pooledh[i] = (j < 300) ? (_Float16)pooled[gi * 300 + j] : (_Float16)0.f;
}

// pooled[256x320] @ fc1[320x64] via WMMA; 16 blocks x 2 waves
__global__ void __launch_bounds__(64) k_fc1(const _Float16* __restrict__ Ah,
                                            const _Float16* __restrict__ Bh,
                                            float* __restrict__ out) {
  int b = blockIdx.x;
  int wave = threadIdx.x >> 5, lane = threadIdx.x & 31;
  for (int it = 0; it < 2; ++it) {
    int col0 = (wave * 2 + it) * 16;
    v8f acc = {};
    for (int ks = 0; ks < FCK / 32; ++ks) {
      v16h a  = wmma_load_a(Ah + ks * 32, b * 16, FCK, lane);
      v16h bf = wmma_load_b(Bh + (size_t)(ks * 32) * FCN, col0, FCN, lane);
      acc = __builtin_amdgcn_wmma_f32_16x16x32_f16(false, a, false, bf, (short)0, acc,
                                                   false, false);
    }
    wmma_store_d(out, acc, b * 16, col0, FCN, lane);
  }
}

__global__ void __launch_bounds__(256) k_fc2(const float* __restrict__ tmp1,
                                             const float* __restrict__ fc1_b,
                                             const float* __restrict__ fc2_w,
                                             const float* __restrict__ fc2_b,
                                             float* __restrict__ out) {
  int t = blockIdx.x * 256 + threadIdx.x;
  if (t >= G * 2) return;
  int g = t >> 1, o = t & 1;
  float acc = fc2_b[o];
  for (int j = 0; j < 50; ++j) {
    float y = tmp1[g * FCN + j] + fc1_b[j];
    y = y > 0.f ? y : 0.f;
    acc += y * fc2_w[j * 2 + o];
  }
  out[g * 2 + o] = acc;
}

// =====================================================================
extern "C" void kernel_launch(void* const* d_in, const int* in_sizes, int n_in,
                              void* d_out, int out_size, void* d_ws, size_t ws_size,
                              hipStream_t stream) {
  (void)in_sizes; (void)n_in; (void)out_size; (void)ws_size;
  const float* x        = (const float*)d_in[0];
  const int*   ei       = (const int*)d_in[1];
  const float* eattr    = (const float*)d_in[2];
  const float* egat_w   = (const float*)d_in[4];
  const float* att_src  = (const float*)d_in[5];
  const float* att_dst  = (const float*)d_in[6];
  const float* att_edge = (const float*)d_in[7];
  const float* gc_wr    = (const float*)d_in[8];
  const float* gc_wn    = (const float*)d_in[9];
  const float* gc_b     = (const float*)d_in[10];
  const float* fc1_w    = (const float*)d_in[11];
  const float* fc1_b    = (const float*)d_in[12];
  const float* fc2_w    = (const float*)d_in[13];
  const float* fc2_b    = (const float*)d_in[14];
  float* out = (float*)d_out;

  // workspace carve-up (256B aligned)
  char* base = (char*)d_ws;
  size_t off = 0;
  auto carve = [&](size_t bytes) {
    char* p = base + off;
    off = (off + bytes + 255) & ~(size_t)255;
    return p;
  };
  float*    ws_h      = (float*)carve((size_t)NTOT * C * sizeof(float));
  float*    ws_ew     = (float*)carve((size_t)ETOT * sizeof(float));
  int*      ws_src    = (int*)carve((size_t)ETOT * sizeof(int));
  int*      ws_start  = (int*)carve((size_t)G * (NPG + 1) * sizeof(int));
  float*    ws_pooled = (float*)carve((size_t)G * 300 * sizeof(float));
  _Float16* ws_ph     = (_Float16*)carve((size_t)G * FCK * sizeof(_Float16));
  _Float16* ws_fc1h   = (_Float16*)carve((size_t)FCK * FCN * sizeof(_Float16));
  float*    ws_tmp1   = (float*)carve((size_t)G * FCN * sizeof(float));

  k_prep_fc1<<<(FCK * FCN + 255) / 256, 256, 0, stream>>>(fc1_w, ws_fc1h);
  k_egat<<<G, 256, 0, stream>>>(x, ei, eattr, egat_w, att_src, att_dst, att_edge,
                                ws_h, ws_ew, ws_src, ws_start, ws_pooled);
  k_gc<<<G, 256, 0, stream>>>(ws_h, gc_wr, gc_wn, gc_b, ws_ew, ws_src, ws_start,
                              ws_pooled);
  k_pool_cvt<<<(G * FCK + 255) / 256, 256, 0, stream>>>(ws_pooled, ws_ph);
  k_fc1<<<G / 16, 64, 0, stream>>>(ws_ph, ws_fc1h, ws_tmp1);
  k_fc2<<<(G * 2 + 255) / 256, 256, 0, stream>>>(ws_tmp1, fc1_b, fc2_w, fc2_b, out);
}
